// MultiHeadAttention_58102317580317
// MI455X (gfx1250) — matrix operand order repaired and checked by reading, not compiled, not run on hardware
//
#include <hip/hip_runtime.h>

// ---------------- problem constants ----------------
constexpr int Dm = 1024;   // d_model
constexpr int Sq = 2048;   // sequence length
constexpr int Bb = 2;      // batch
constexpr int Hh = 16;     // heads
constexpr int DK = 64;     // head dim
constexpr int Mrows = Bb * Sq; // 4096

typedef _Float16 h16 __attribute__((ext_vector_type(16)));
typedef _Float16 h8  __attribute__((ext_vector_type(8)));
typedef float    f8  __attribute__((ext_vector_type(8)));
typedef unsigned int u32x4 __attribute__((ext_vector_type(4)));
typedef int          i32x8 __attribute__((ext_vector_type(8)));
typedef int          i32x4 __attribute__((ext_vector_type(4)));

__device__ __forceinline__ h16 cat16(h8 lo, h8 hi) {
  return __builtin_shufflevector(lo, hi, 0,1,2,3,4,5,6,7,8,9,10,11,12,13,14,15);
}
__device__ __forceinline__ h8 ld8(const _Float16* p) {
  return *(const h8*)p;
}

// A-fragment (16x32 f16): lane holds row m=lane&15.
// lanes 0-15 hold k 0..15 contiguous, lanes 16-31 hold k 16..31 contiguous.
__device__ __forceinline__ h16 a_frag(const _Float16* base, int stride, int koff, int lane) {
  const _Float16* row = base + (size_t)(lane & 15) * stride + koff + ((lane >> 4) << 4);
  return cat16(ld8(row), ld8(row + 8));
}

// B-fragment (32x16 f16): lane holds column n=lane&15.
// lanes 0-15 hold k 0..15 contiguous, lanes 16-31 hold k 16..31 contiguous.
__device__ __forceinline__ h16 b_frag(const _Float16* base, size_t stride, int koff, int lane) {
  const _Float16* col = base + (size_t)(lane & 15) * stride + koff + ((lane >> 4) << 4);
  return cat16(ld8(col), ld8(col + 8));
}

__device__ __forceinline__ f8 wmma_f16(h16 a, h16 b, f8 c) {
  return __builtin_amdgcn_wmma_f32_16x16x32_f16(false, a, false, b, (short)0, c, false, false);
}

// ---------------------------------------------------------------------------
// Generic GEMM: C[Mrows x Dm] = A[Mrows x Dm] @ W[Dm x Dm] + bias
// Block tile 128x128, 8 waves (256 thr), wave tile 32x64, k-step 32.
// MODE 0: out f16, head-major  [B][H][S][DK]   (Q and K)
// MODE 2: out f16, transposed  [B][H][DK][S]   (V, ready as WMMA B-frags)
// MODE 3: out fp32, row-major  [row][col]      (final projection)
// ---------------------------------------------------------------------------
template<int MODE, bool A32>
__global__ __launch_bounds__(256) void gemm_kernel(const void* __restrict__ Ain,
                                                   const float* __restrict__ W,
                                                   const float* __restrict__ bias,
                                                   void* __restrict__ Out) {
  __shared__ __align__(16) _Float16 lA[128][40];
  __shared__ __align__(16) _Float16 lB[128][40];   // stored transposed: lB[n][k]

  const int tid  = threadIdx.x;
  const int lane = tid & 31;
  const int wid  = tid >> 5;
  const int waveM = wid >> 1;
  const int waveN = wid & 1;
  const int m0 = blockIdx.y * 128;
  const int n0 = blockIdx.x * 128;

  f8 acc[2][4];
  for (int i = 0; i < 2; ++i)
    for (int j = 0; j < 4; ++j)
      acc[i][j] = f8{};

  for (int k0 = 0; k0 < Dm; k0 += 32) {
    if (A32) {
      const float* A = (const float*)Ain;
      #pragma unroll
      for (int i = 0; i < 4; ++i) {
        int idx = tid + i * 256;
        int r  = idx >> 3;
        int c4 = idx & 7;
        float4 v = *(const float4*)(A + (size_t)(m0 + r) * Dm + k0 + c4 * 4);
        _Float16* dst = &lA[r][c4 * 4];
        dst[0] = (_Float16)v.x; dst[1] = (_Float16)v.y;
        dst[2] = (_Float16)v.z; dst[3] = (_Float16)v.w;
      }
    } else {
      const _Float16* A = (const _Float16*)Ain;
      #pragma unroll
      for (int i = 0; i < 2; ++i) {
        int idx = tid + i * 256;
        int r  = idx >> 2;
        int c8 = idx & 3;
        *(h8*)&lA[r][c8 * 8] = ld8(A + (size_t)(m0 + r) * Dm + k0 + c8 * 8);
      }
    }
    #pragma unroll
    for (int i = 0; i < 4; ++i) {
      int idx = tid + i * 256;
      int kr = idx >> 5;
      int c4 = idx & 31;
      float4 v = *(const float4*)(W + (size_t)(k0 + kr) * Dm + n0 + c4 * 4);
      lB[c4 * 4 + 0][kr] = (_Float16)v.x;
      lB[c4 * 4 + 1][kr] = (_Float16)v.y;
      lB[c4 * 4 + 2][kr] = (_Float16)v.z;
      lB[c4 * 4 + 3][kr] = (_Float16)v.w;
    }
    __syncthreads();

    h16 af0 = a_frag(&lA[waveM * 32 + 0][0],  40, 0, lane);
    h16 af1 = a_frag(&lA[waveM * 32 + 16][0], 40, 0, lane);
    #pragma unroll
    for (int nt = 0; nt < 4; ++nt) {
      h16 bf = b_frag(&lB[waveN * 64 + nt * 16][0], 40, 0, lane);
      acc[0][nt] = wmma_f16(af0, bf, acc[0][nt]);
      acc[1][nt] = wmma_f16(af1, bf, acc[1][nt]);
    }
    __syncthreads();
  }

  #pragma unroll
  for (int mt = 0; mt < 2; ++mt)
    #pragma unroll
    for (int nt = 0; nt < 4; ++nt)
      #pragma unroll
      for (int r = 0; r < 8; ++r) {
        int row = m0 + waveM * 32 + mt * 16 + r + ((lane >> 4) << 3);
        int col = n0 + waveN * 64 + nt * 16 + (lane & 15);
        float v = acc[mt][nt][r] + bias[col];
        if (MODE == 3) {
          ((float*)Out)[(size_t)row * Dm + col] = v;
        } else {
          int b  = row >> 11;
          int s  = row & (Sq - 1);
          int hd = col >> 6;
          int dk = col & (DK - 1);
          _Float16 hv = (_Float16)v;
          if (MODE == 2)
            ((_Float16*)Out)[(((size_t)(b * Hh + hd)) * DK + dk) * Sq + s] = hv;
          else
            ((_Float16*)Out)[(((size_t)(b * Hh + hd)) * Sq + s) * DK + dk] = hv;
        }
      }
}

// ---------------------------------------------------------------------------
// Tensor Data Mover staging (CDNA5): descriptor-driven global->LDS DMA.
// This toolchain ships the 6-arg builtin:
//   (uint32x4 g0, int32x8 g1, int32x4 g2, int32x4 g3, int32x8 extra, i32 cpol)
// Gated on builtin availability; falls back to manual global->LDS staging.
// ---------------------------------------------------------------------------
#if __has_builtin(__builtin_amdgcn_tensor_load_to_lds) && __has_builtin(__builtin_amdgcn_s_wait_tensorcnt)
#define ATTN_TDM 1
#else
#define ATTN_TDM 0
#endif

#if ATTN_TDM
// group1 word0: data_size=2B (1<<16), pad_enable (1<<20), pad_interval, pad_amount
// K tile: 1D run of 2048 halves; pad 4 DWORDs after every 32 DWORDs -> LDS row stride 72 halves
constexpr unsigned TDM_W0_K = (1u << 16) | (1u << 20) | (4u << 22) | (3u << 25);
// V tile: 64 rows x 32 halves; pad 4 DWORDs after every 16 DWORDs -> LDS row stride 40 halves
constexpr unsigned TDM_W0_V = (1u << 16) | (1u << 20) | (3u << 22) | (3u << 25);

__device__ __forceinline__ void tdm_load_2d(unsigned lds_addr, const void* gaddr,
                                            unsigned word0,
                                            unsigned tensor_dim0, unsigned tensor_dim1,
                                            unsigned tile_dim0, unsigned tile_dim1,
                                            unsigned long long stride0) {
  unsigned long long ga = (unsigned long long)(size_t)gaddr;
  u32x4 g0;
  g0[0] = 1u;                                            // count=1, user mode, no gather
  g0[1] = lds_addr;                                      // LDS byte address
  g0[2] = (unsigned)(ga & 0xffffffffu);                  // global_addr[31:0]
  g0[3] = (unsigned)((ga >> 32) & 0x1ffffffu) | (2u << 30);  // global_addr[56:32] | type=2
  i32x8 g1;
  g1[0] = (int)word0;
  g1[1] = (int)((tensor_dim0 & 0xffffu) << 16);          // atomic_barrier_addr=0 | dim0 lo
  g1[2] = (int)((tensor_dim0 >> 16) | ((tensor_dim1 & 0xffffu) << 16));
  g1[3] = (int)((tensor_dim1 >> 16) | (tile_dim0 << 16));
  g1[4] = (int)(tile_dim1 & 0xffffu);                    // tile_dim2 = 0
  g1[5] = (int)(stride0 & 0xffffffffu);                  // tensor_dim0_stride lo
  g1[6] = (int)((stride0 >> 32) & 0xffffu);              // stride hi | dim1_stride lo = 0
  g1[7] = 0;
  i32x4 z4 = {0, 0, 0, 0};
  i32x8 z8 = {0, 0, 0, 0, 0, 0, 0, 0};
  __builtin_amdgcn_tensor_load_to_lds(g0, g1, z4, z4, z8, 0);
}

__device__ __forceinline__ void tdm_issue(unsigned ldsK, unsigned ldsV,
                                          const _Float16* Kg, const _Float16* Vg) {
  // K: 32 rows x 64 halves, rows contiguous -> 1D copy of 2048 halves
  tdm_load_2d(ldsK, Kg, TDM_W0_K, 2048u, 1u, 2048u, 0u, 2048ull);
  // V: 64 d-rows x 32 key-cols, global row stride Sq halves
  tdm_load_2d(ldsV, Vg, TDM_W0_V, 2048u, 64u, 32u, 64u, (unsigned long long)Sq);
}
#endif

// ---------------------------------------------------------------------------
// Fused flash attention: block = 4 waves x 32 q-rows = 128 q-rows of one (b,h).
// K/V tiles (32 keys) staged in LDS, shared by all waves, TDM double-buffered.
// ---------------------------------------------------------------------------
__global__ __launch_bounds__(128) void attn_kernel(const _Float16* __restrict__ Qh,
                                                   const _Float16* __restrict__ Kh,
                                                   const _Float16* __restrict__ Vt,
                                                   const int* __restrict__ mask,
                                                   _Float16* __restrict__ Xh) {
  __shared__ __align__(16) _Float16 lK[2][32][72];  // [buf][key][d], padded rows (bank-tiling)
  __shared__ __align__(16) _Float16 lV[2][64][40];  // [buf][d][key], padded rows
  __shared__ __align__(16) _Float16 lP[4][16][40];  // per-wave P reshape tile

  const int tid  = threadIdx.x;
  const int lane = tid & 31;
  const int wid  = tid >> 5;
  const int bh   = blockIdx.y;
  const int b    = bh >> 4;
  const int hh   = bh & (Hh - 1);
  const int q0   = blockIdx.x * 128 + wid * 32;

  const _Float16* Qbase = Qh + ((size_t)bh * Sq + q0) * DK;
  const _Float16* Kbase = Kh + (size_t)bh * Sq * DK;
  const _Float16* Vbase = Vt + (size_t)bh * DK * Sq;
  const int* mbase = mask + (size_t)b * Sq * Sq;

  // Q fragments: 2 m-tiles x 2 k-chunks, loaded once
  h16 qa[2][2];
  qa[0][0] = a_frag(Qbase,           DK, 0,  lane);
  qa[0][1] = a_frag(Qbase,           DK, 32, lane);
  qa[1][0] = a_frag(Qbase + 16 * DK, DK, 0,  lane);
  qa[1][1] = a_frag(Qbase + 16 * DK, DK, 32, lane);

  f8 O[2][4];
  #pragma unroll
  for (int mt = 0; mt < 2; ++mt)
    #pragma unroll
    for (int t = 0; t < 4; ++t) O[mt][t] = f8{};
  float mrun[2][8], lrun[2][8];
  #pragma unroll
  for (int mt = 0; mt < 2; ++mt)
    #pragma unroll
    for (int r = 0; r < 8; ++r) { mrun[mt][r] = -__builtin_inff(); lrun[mt][r] = 0.0f; }

  constexpr int NB = Sq / 32;   // 64 key blocks

#if ATTN_TDM
  if (wid == 0)
    tdm_issue((unsigned)(size_t)&lK[0][0][0], (unsigned)(size_t)&lV[0][0][0], Kbase, Vbase);
#endif

  for (int jb = 0; jb < NB; ++jb) {
    const int j   = jb * 32;
    const int buf = jb & 1;

#if ATTN_TDM
    if (wid == 0) {
      if (jb + 1 < NB) {
        tdm_issue((unsigned)(size_t)&lK[buf ^ 1][0][0], (unsigned)(size_t)&lV[buf ^ 1][0][0],
                  Kbase + (size_t)(j + 32) * DK, Vbase + (j + 32));
        __builtin_amdgcn_s_wait_tensorcnt(2);   // current tile pair done; next pair in flight
      } else {
        __builtin_amdgcn_s_wait_tensorcnt(0);
      }
    }
    __syncthreads();
#else
    // manual cooperative staging (same LDS layout)
    #pragma unroll
    for (int i = 0; i < 2; ++i) {
      int idx = tid + i * 128;                  // 0..255
      int r  = idx >> 3;                        // key 0..31
      int c8 = idx & 7;
      *(h8*)&lK[buf][r][c8 * 8] = ld8(Kbase + (size_t)(j + r) * DK + c8 * 8);
    }
    #pragma unroll
    for (int i = 0; i < 2; ++i) {
      int idx = tid + i * 128;                  // 0..255
      int r  = idx >> 2;                        // d 0..63
      int c8 = idx & 3;
      *(h8*)&lV[buf][r][c8 * 8] = ld8(Vbase + (size_t)r * Sq + j + c8 * 8);
    }
    __syncthreads();
#endif

    // K/V fragments from LDS (shared across waves, conflict-free padded strides)
    h16 kb[2][2], vb[4];
    kb[0][0] = b_frag(&lK[buf][0][0],  72, 0,  lane);
    kb[0][1] = b_frag(&lK[buf][0][0],  72, 32, lane);
    kb[1][0] = b_frag(&lK[buf][16][0], 72, 0,  lane);
    kb[1][1] = b_frag(&lK[buf][16][0], 72, 32, lane);
    #pragma unroll
    for (int t = 0; t < 4; ++t)
      vb[t] = b_frag(&lV[buf][t * 16][0], 40, 0, lane);

    #pragma unroll
    for (int mt = 0; mt < 2; ++mt) {
      f8 s0 = f8{}, s1 = f8{};
      s0 = wmma_f16(qa[mt][0], kb[0][0], s0);
      s0 = wmma_f16(qa[mt][1], kb[0][1], s0);
      s1 = wmma_f16(qa[mt][0], kb[1][0], s1);
      s1 = wmma_f16(qa[mt][1], kb[1][1], s1);

      float p0[8], p1[8], alpha[8];
      #pragma unroll
      for (int r = 0; r < 8; ++r) {
        int qrow = q0 + mt * 16 + r + ((lane >> 4) << 3);
        int kc0  = j + (lane & 15);
        float v0 = s0[r] * 0.125f;              // 1/sqrt(DK)
        float v1 = s1[r] * 0.125f;
        if (mbase[(size_t)qrow * Sq + kc0] == 0)      v0 = -__builtin_inff();
        if (mbase[(size_t)qrow * Sq + kc0 + 16] == 0) v1 = -__builtin_inff();
        float mx = fmaxf(v0, v1);
        mx = fmaxf(mx, __shfl_xor(mx, 1));
        mx = fmaxf(mx, __shfl_xor(mx, 2));
        mx = fmaxf(mx, __shfl_xor(mx, 4));
        mx = fmaxf(mx, __shfl_xor(mx, 8));
        float mnew = fmaxf(mrun[mt][r], mx);
        float a    = __expf(mrun[mt][r] - mnew);
        p0[r] = __expf(v0 - mnew);
        p1[r] = __expf(v1 - mnew);
        float ps = p0[r] + p1[r];
        ps += __shfl_xor(ps, 1);
        ps += __shfl_xor(ps, 2);
        ps += __shfl_xor(ps, 4);
        ps += __shfl_xor(ps, 8);
        lrun[mt][r] = lrun[mt][r] * a + ps;
        mrun[mt][r] = mnew;
        alpha[r] = a;
      }

      #pragma unroll
      for (int t = 0; t < 4; ++t)
        #pragma unroll
        for (int r = 0; r < 8; ++r)
          O[mt][t][r] *= alpha[r];

      // reshape P (C-layout) -> A-fragment layout via per-wave LDS tile
      #pragma unroll
      for (int r = 0; r < 8; ++r) {
        int m = r + ((lane >> 4) << 3);
        lP[wid][m][lane & 15]        = (_Float16)p0[r];
        lP[wid][m][16 + (lane & 15)] = (_Float16)p1[r];
      }
      // same-wave DS ops are in-order: no barrier needed
      h16 pa = a_frag(&lP[wid][0][0], 40, 0, lane);

      #pragma unroll
      for (int t = 0; t < 4; ++t)
        O[mt][t] = wmma_f16(pa, vb[t], O[mt][t]);
    }
    __syncthreads();
  }

  // finalize: divide by row sums, scatter to [B][S][H*DK] f16
  #pragma unroll
  for (int mt = 0; mt < 2; ++mt)
    #pragma unroll
    for (int r = 0; r < 8; ++r) {
      int qrow  = q0 + mt * 16 + r + ((lane >> 4) << 3);
      float inv = 1.0f / lrun[mt][r];
      #pragma unroll
      for (int t = 0; t < 4; ++t) {
        int dcol = hh * DK + t * 16 + (lane & 15);
        Xh[((size_t)b * Sq + qrow) * Dm + dcol] = (_Float16)(O[mt][t][r] * inv);
      }
    }
}

// ---------------------------------------------------------------------------
extern "C" void kernel_launch(void* const* d_in, const int* in_sizes, int n_in,
                              void* d_out, int out_size, void* d_ws, size_t ws_size,
                              hipStream_t stream) {
  const float* q    = (const float*)d_in[0];
  const float* k    = (const float*)d_in[1];
  const float* v    = (const float*)d_in[2];
  const int*   mask = (const int*)d_in[3];
  const float* Wq = (const float*)d_in[4];  const float* bq = (const float*)d_in[5];
  const float* Wk = (const float*)d_in[6];  const float* bk = (const float*)d_in[7];
  const float* Wv = (const float*)d_in[8];  const float* bv = (const float*)d_in[9];
  const float* Wo = (const float*)d_in[10]; const float* bo = (const float*)d_in[11];

  char* ws = (char*)d_ws;
  const size_t BUF = (size_t)Mrows * Dm * sizeof(_Float16);   // 8 MB each
  _Float16* Qh = (_Float16*)(ws + 0 * BUF);   // [B][H][S][DK]
  _Float16* Kh = (_Float16*)(ws + 1 * BUF);   // [B][H][S][DK]
  _Float16* Vt = (_Float16*)(ws + 2 * BUF);   // [B][H][DK][S]
  _Float16* Xh = (_Float16*)(ws + 3 * BUF);   // [B][S][H*DK]

  dim3 gemm_grid(Dm / 128, Mrows / 128);      // (8, 32)
  gemm_kernel<0, true><<<gemm_grid, 256, 0, stream>>>(q, Wq, bq, Qh);
  gemm_kernel<0, true><<<gemm_grid, 256, 0, stream>>>(k, Wk, bk, Kh);
  gemm_kernel<2, true><<<gemm_grid, 256, 0, stream>>>(v, Wv, bv, Vt);

  dim3 attn_grid(Sq / 128, Bb * Hh);          // (16, 32)
  attn_kernel<<<attn_grid, 128, 0, stream>>>(Qh, Kh, Vt, mask, Xh);

  gemm_kernel<3, false><<<gemm_grid, 256, 0, stream>>>(Xh, Wo, bo, (float*)d_out);
}
